// MambaEncoderLayer_73821897883826
// MI455X (gfx1250) — compile-verified, matrix operand order checked by reading, not run on hardware
//
#include <hip/hip_runtime.h>
#include <hip/hip_bf16.h>

typedef __attribute__((ext_vector_type(16))) _Float16 v16h;
typedef __attribute__((ext_vector_type(8)))  _Float16 v8h;
typedef __attribute__((ext_vector_type(8)))  float    v8f;

#define LSEQ 4096
#define CCH  128
#define NBATCH 2
#define BB   6              // 3 streams * 2 batches
#define M1   (BB*LSEQ)      // 24576 rows through the mamba mixer
#define M2   (NBATCH*LSEQ)  // 8192 rows through the FFN
#define DI   384
#define NST  4
#define RRANK 128
#define FCH  512
#define NXP  136            // R + 2N
#define NCHUNK 64
#define CLEN   64

enum { EPI_NONE=0, EPI_BIAS_SOFTPLUS=1, EPI_BIAS=2, EPI_BIAS_RESID=3 };

// ---------------- device helpers ----------------
__device__ __forceinline__ float wave_sum(float v) {
    for (int m = 16; m > 0; m >>= 1) v += __shfl_xor(v, m, 32);
    return v;
}
__device__ __forceinline__ float silu_f(float x) { return x / (1.f + __expf(-x)); }

// generic-AS pointer -> LDS byte offset (for async-to-LDS VDST operand)
__device__ __forceinline__ unsigned lds_off_of(const void* p) {
    return (unsigned)(size_t)(__attribute__((address_space(3))) const char*)p;
}

// =====================================================================
// WMMA GEMM, C = epi(A(MxK,f16) * W(NxK,f16)^T)
//  - block = 256 threads = 8 waves; block computes 128(M) x 64(N)
//  - B panel (64 rows of W, i.e. 64 output columns) staged in LDS via
//    gfx1250 async-to-LDS loads; rows padded by 8 halfs to avoid the
//    256B bank stride
//  - each wave: 16x64 output, one A fragment feeds 4 WMMAs per K-step
//  - out-of-range N columns compute garbage and are masked at the store
// =====================================================================
template<int K, int EPI>
__global__ __launch_bounds__(256)
void k_gemm_wmma(const _Float16* __restrict__ A, const _Float16* __restrict__ W,
                 float* __restrict__ C, int M, int N, int ldc,
                 const float* __restrict__ bias, const float* __restrict__ resid,
                 int ngroups)
{
    constexpr int KP = K + 8;                       // padded LDS row, in halfs
    __shared__ __align__(16) _Float16 bpanel[64 * KP];

    const int ng = blockIdx.x % ngroups;
    const int mg = blockIdx.x / ngroups;
    const int n0 = ng * 64;

    // ---- stage B panel (<=64 x K halfs) into LDS with async copies ----
    {
        const int chunksPerRow = K / 8;             // 16-byte chunks per row
        int rowsValid = N - n0; if (rowsValid > 64) rowsValid = 64;
        const int total = rowsValid * chunksPerRow;
        for (int cidx = threadIdx.x; cidx < total; cidx += 256) {
            int r = cidx / chunksPerRow, cc = cidx - r * chunksPerRow;
            unsigned lo = lds_off_of(&bpanel[r * KP + cc * 8]);
            const _Float16* gp = W + (size_t)(n0 + r) * K + cc * 8;
            asm volatile("global_load_async_to_lds_b128 %0, %1, off"
                         :: "v"(lo), "v"(gp) : "memory");
        }
        asm volatile("s_wait_asynccnt 0x0" ::: "memory");
    }
    __syncthreads();

    const int wid  = threadIdx.x >> 5;
    const int lane = threadIdx.x & 31;
    const int lm   = lane & 15;
    const int khalf = lane >> 4;                    // K[0..7]/K[16..23] vs K[8..15]/K[24..31]
    const int mt = mg * 8 + wid;                    // this wave's 16-row tile (M % 128 == 0)

    const _Float16* arow = A + (size_t)(mt * 16 + lm) * K + khalf * 8;
    const _Float16* lrow = &bpanel[lm * KP + khalf * 8];

    v8f acc[4] = {{}, {}, {}, {}};
    for (int k0 = 0; k0 < K; k0 += 32) {
        v8h alo = *(const v8h*)(arow + k0);
        v8h ahi = *(const v8h*)(arow + k0 + 16);
        v16h av = __builtin_shufflevector(alo, ahi, 0,1,2,3,4,5,6,7,8,9,10,11,12,13,14,15);
        #pragma unroll
        for (int j = 0; j < 4; ++j) {
            const _Float16* bp = lrow + j * 16 * KP + k0;
            v8h blo = *(const v8h*)(bp);
            v8h bhi = *(const v8h*)(bp + 16);
            v16h bv = __builtin_shufflevector(blo, bhi, 0,1,2,3,4,5,6,7,8,9,10,11,12,13,14,15);
            acc[j] = __builtin_amdgcn_wmma_f32_16x16x32_f16(false, av, false, bv,
                                                            (short)0, acc[j], false, false);
        }
    }

    // D layout: VGPR r holds rows M=r (lanes 0-15) and M=r+8 (lanes 16-31)
    #pragma unroll
    for (int j = 0; j < 4; ++j) {
        int no = n0 + j * 16 + lm;
        if (no >= N) continue;
        #pragma unroll
        for (int r = 0; r < 8; ++r) {
            int mo = mt * 16 + r + khalf * 8;
            float v = acc[j][r];
            if (EPI == EPI_BIAS_SOFTPLUS) { v += bias[no]; v = (v > 20.f) ? v : log1pf(__expf(v)); }
            else if (EPI == EPI_BIAS)      { v += bias[no]; }
            else if (EPI == EPI_BIAS_RESID){ v += bias[no] + resid[(size_t)mo * ldc + no]; }
            C[(size_t)mo * ldc + no] = v;
        }
    }
}

// ---------------- misc small kernels ----------------
__global__ void k_f32_to_f16(const float* __restrict__ s, _Float16* __restrict__ d, int n) {
    int i = blockIdx.x * blockDim.x + threadIdx.x;
    if (i < n) d[i] = (_Float16)s[i];
}

__global__ void k_invperm(const int* __restrict__ p, int* __restrict__ inv) {
    int i = blockIdx.x * blockDim.x + threadIdx.x;
    if (i < LSEQ) inv[p[i]] = i;
}

// build x_in = [x, flip(x), x[:,perm]] and layernorm -> f16 rows (one wave per row)
__global__ __launch_bounds__(128)
void k_prep_ln(const float* __restrict__ x, const int* __restrict__ ridx,
               const float* __restrict__ g, const float* __restrict__ bta,
               _Float16* __restrict__ xn)
{
    int wave = blockIdx.x * 4 + (threadIdx.x >> 5);
    int lane = threadIdx.x & 31;
    if (wave >= M1) return;
    int t = wave % LSEQ, bb = wave / LSEQ;
    int strm = bb / NBATCH, b = bb % NBATCH;
    int ts = (strm == 0) ? t : (strm == 1) ? (LSEQ - 1 - t) : ridx[t];
    const float* row = x + ((size_t)b * LSEQ + ts) * CCH;
    float v[4]; float s1 = 0.f, s2 = 0.f;
    for (int i = 0; i < 4; ++i) { float xv = row[lane * 4 + i]; v[i] = xv; s1 += xv; s2 += xv * xv; }
    s1 = wave_sum(s1); s2 = wave_sum(s2);
    float mean = s1 * (1.f / CCH);
    float var  = s2 * (1.f / CCH) - mean * mean;
    float rs = rsqrtf(var + 1e-5f);
    _Float16* orow = xn + (size_t)wave * CCH;
    for (int i = 0; i < 4; ++i) {
        int c = lane * 4 + i;
        orow[c] = (_Float16)((v[i] - mean) * rs * g[c] + bta[c]);
    }
}

// causal depthwise conv K=4 over L + bias + silu; h = hg[:, 0:384]
__global__ void k_conv_silu(const float* __restrict__ hg, const float* __restrict__ cw,
                            const float* __restrict__ cb,
                            float* __restrict__ u, _Float16* __restrict__ u16)
{
    int idx = blockIdx.x * blockDim.x + threadIdx.x;
    if (idx >= M1 * DI) return;
    int d = idx % DI, r = idx / DI;
    int t = r % LSEQ, rbase = r - t;
    float acc = cb[d];
    #pragma unroll
    for (int j = 0; j < 4; ++j) {
        int tj = t + j - 3;
        if (tj >= 0) acc += cw[d * 4 + j] * hg[(size_t)(rbase + tj) * 768 + d];
    }
    float s = silu_f(acc);
    u[idx] = s;
    u16[idx] = (_Float16)s;
}

// dt input slice (cols 0..127 of dtBC) -> f16
__global__ void k_dtcvt(const float* __restrict__ dtbc, _Float16* __restrict__ d16) {
    int i = blockIdx.x * blockDim.x + threadIdx.x;
    if (i >= M1 * RRANK) return;
    int c = i % RRANK, r = i / RRANK;
    d16[i] = (_Float16)dtbc[(size_t)r * NXP + c];
}

// -------- chunked linear scan: s_t = dA_t*s + dt_t*B_t*u_t, y_t = <s, C_t> --------
__global__ void k_scan_pass1(const float* __restrict__ dt, const float* __restrict__ dtbc,
                             const float* __restrict__ u, const float* __restrict__ alog,
                             float* __restrict__ Ap, float* __restrict__ Sc)
{
    int idx = blockIdx.x * blockDim.x + threadIdx.x;
    if (idx >= BB * NCHUNK * DI) return;
    int d = idx % DI;
    int c = (idx / DI) % NCHUNK;
    int bb = idx / (DI * NCHUNK);
    float An[NST];
    #pragma unroll
    for (int n = 0; n < NST; ++n) An[n] = -__expf(alog[d * NST + n]);
    float ap[NST] = {1.f,1.f,1.f,1.f}, s[NST] = {0.f,0.f,0.f,0.f};
    int t0 = c * CLEN;
    for (int tt = 0; tt < CLEN; ++tt) {
        size_t rr = (size_t)bb * LSEQ + t0 + tt;
        float dtv = dt[rr * DI + d];
        float uv  = u [rr * DI + d];
        const float* bs = dtbc + rr * NXP + RRANK;       // B_t[0..3]
        #pragma unroll
        for (int n = 0; n < NST; ++n) {
            float dA = __expf(dtv * An[n]);
            s[n]  = dA * s[n] + dtv * bs[n] * uv;
            ap[n] *= dA;
        }
    }
    size_t o = (((size_t)bb * DI + d) * NCHUNK + c) * NST;
    #pragma unroll
    for (int n = 0; n < NST; ++n) { Ap[o + n] = ap[n]; Sc[o + n] = s[n]; }
}

__global__ void k_scan_pass2(const float* __restrict__ Ap, const float* __restrict__ Sc,
                             float* __restrict__ initS)
{
    int idx = blockIdx.x * blockDim.x + threadIdx.x;
    if (idx >= BB * DI * NST) return;
    int n = idx % NST;
    int d = (idx / NST) % DI;
    int bb = idx / (NST * DI);
    size_t base = ((size_t)bb * DI + d) * NCHUNK * NST + n;
    float run = 0.f;
    for (int c = 0; c < NCHUNK; ++c) {
        size_t o = base + (size_t)c * NST;
        initS[o] = run;
        run = Ap[o] * run + Sc[o];
    }
}

__global__ void k_scan_pass3(const float* __restrict__ dt, const float* __restrict__ dtbc,
                             const float* __restrict__ u, const float* __restrict__ alog,
                             const float* __restrict__ initS, const float* __restrict__ hg,
                             const float* __restrict__ Dp, _Float16* __restrict__ y16)
{
    int idx = blockIdx.x * blockDim.x + threadIdx.x;
    if (idx >= BB * NCHUNK * DI) return;
    int d = idx % DI;
    int c = (idx / DI) % NCHUNK;
    int bb = idx / (DI * NCHUNK);
    float An[NST];
    #pragma unroll
    for (int n = 0; n < NST; ++n) An[n] = -__expf(alog[d * NST + n]);
    float s[NST];
    size_t o = (((size_t)bb * DI + d) * NCHUNK + c) * NST;
    #pragma unroll
    for (int n = 0; n < NST; ++n) s[n] = initS[o + n];
    float dpd = Dp[d];
    int t0 = c * CLEN;
    for (int tt = 0; tt < CLEN; ++tt) {
        size_t rr = (size_t)bb * LSEQ + t0 + tt;
        float dtv = dt[rr * DI + d];
        float uv  = u [rr * DI + d];
        const float* bs = dtbc + rr * NXP + RRANK;        // B_t
        const float* cs = dtbc + rr * NXP + RRANK + NST;  // C_t
        float y = 0.f;
        #pragma unroll
        for (int n = 0; n < NST; ++n) {
            float dA = __expf(dtv * An[n]);
            s[n] = dA * s[n] + dtv * bs[n] * uv;
            y += s[n] * cs[n];
        }
        float gate = hg[rr * 768 + DI + d];
        y16[rr * DI + d] = (_Float16)((y + uv * dpd) * silu_f(gate));
    }
}

// mean over L per (stream,batch,channel) — flip/permutation invariant
__global__ __launch_bounds__(128)
void k_gate_mean(const float* __restrict__ yout, float* __restrict__ gmean)
{
    int wave = blockIdx.x * 4 + (threadIdx.x >> 5);
    int lane = threadIdx.x & 31;
    if (wave >= BB * CCH) return;
    int c = wave % CCH, bb = wave / CCH;
    float s = 0.f;
    for (int t = lane; t < LSEQ; t += 32) s += yout[((size_t)bb * LSEQ + t) * CCH + c];
    s = wave_sum(s);
    if (lane == 0) {
        int strm = bb / NBATCH, b = bb % NBATCH;
        gmean[b * (3 * CCH) + strm * CCH + c] = s * (1.f / LSEQ);
    }
}

__global__ __launch_bounds__(96)
void k_gate_softmax(const float* __restrict__ gmean, const float* __restrict__ gw,
                    float* __restrict__ gcoef)
{
    int b = blockIdx.x;
    int j = threadIdx.x >> 5, lane = threadIdx.x & 31;
    __shared__ float lg[3];
    float s = 0.f;
    for (int i = lane; i < 3 * CCH; i += 32) s += gmean[b * 3 * CCH + i] * gw[j * 3 * CCH + i];
    s = wave_sum(s);
    if (lane == 0) lg[j] = s;
    __syncthreads();
    if (threadIdx.x == 0) {
        float m = fmaxf(lg[0], fmaxf(lg[1], lg[2]));
        float e0 = __expf(lg[0] - m), e1 = __expf(lg[1] - m), e2 = __expf(lg[2] - m);
        float inv = 1.f / (e0 + e1 + e2);
        gcoef[b * 3 + 0] = e0 * inv; gcoef[b * 3 + 1] = e1 * inv; gcoef[b * 3 + 2] = e2 * inv;
    }
}

// combine gated streams + residual -> x1, then layernorm2 -> f16 (one wave per row)
__global__ __launch_bounds__(128)
void k_combine_ln2(const float* __restrict__ x, const float* __restrict__ yout,
                   const int* __restrict__ inv, const float* __restrict__ gcoef,
                   const float* __restrict__ g2, const float* __restrict__ b2,
                   float* __restrict__ x1, _Float16* __restrict__ xn2)
{
    int wave = blockIdx.x * 4 + (threadIdx.x >> 5);
    int lane = threadIdx.x & 31;
    if (wave >= M2) return;
    int t = wave % LSEQ, b = wave / LSEQ;
    float g0 = gcoef[b * 3 + 0], g1 = gcoef[b * 3 + 1], g2c = gcoef[b * 3 + 2];
    const float* fr = yout + ((size_t)(0 * NBATCH + b) * LSEQ + t) * CCH;
    const float* rv = yout + ((size_t)(1 * NBATCH + b) * LSEQ + (LSEQ - 1 - t)) * CCH;
    const float* sh = yout + ((size_t)(2 * NBATCH + b) * LSEQ + inv[t]) * CCH;
    const float* xr = x + (size_t)wave * CCH;
    float v[4]; float s1 = 0.f, s2 = 0.f;
    for (int i = 0; i < 4; ++i) {
        int c = lane * 4 + i;
        float vv = xr[c] + g0 * fr[c] + g1 * rv[c] + g2c * sh[c];
        v[i] = vv; s1 += vv; s2 += vv * vv;
    }
    s1 = wave_sum(s1); s2 = wave_sum(s2);
    float mean = s1 * (1.f / CCH);
    float var  = s2 * (1.f / CCH) - mean * mean;
    float rs = rsqrtf(var + 1e-5f);
    float* x1r = x1 + (size_t)wave * CCH;
    _Float16* o2 = xn2 + (size_t)wave * CCH;
    for (int i = 0; i < 4; ++i) {
        int c = lane * 4 + i;
        x1r[c] = v[i];
        o2[c] = (_Float16)((v[i] - mean) * rs * g2[c] + b2[c]);
    }
}

// depthwise 3x3 conv (pad 1) + bias + exact gelu -> f16
__global__ void k_dwconv_gelu(const float* __restrict__ t1, const float* __restrict__ pw,
                              const float* __restrict__ pb, _Float16* __restrict__ t2)
{
    int idx = blockIdx.x * blockDim.x + threadIdx.x;
    if (idx >= M2 * FCH) return;
    int f = idx % FCH, r = idx / FCH;
    int l = r % LSEQ, b = r / LSEQ;
    int h = l >> 6, w = l & 63;
    float acc = pb[f];
    #pragma unroll
    for (int dy = 0; dy < 3; ++dy) {
        int hy = h + dy - 1; if (hy < 0 || hy > 63) continue;
        #pragma unroll
        for (int dx = 0; dx < 3; ++dx) {
            int wx = w + dx - 1; if (wx < 0 || wx > 63) continue;
            acc += pw[f * 9 + dy * 3 + dx] * t1[((size_t)b * LSEQ + hy * 64 + wx) * FCH + f];
        }
    }
    t2[idx] = (_Float16)(0.5f * acc * (1.f + erff(acc * 0.70710678118f)));
}

// ---------------- host orchestration ----------------
extern "C" void kernel_launch(void* const* d_in, const int* in_sizes, int n_in,
                              void* d_out, int out_size, void* d_ws, size_t ws_size,
                              hipStream_t stream) {
    (void)in_sizes; (void)n_in; (void)out_size; (void)ws_size;
    const float* x     = (const float*)d_in[0];
    const int*   ridx  = (const int*)  d_in[1];
    const float* n1g   = (const float*)d_in[4];
    const float* n1b   = (const float*)d_in[5];
    const float* inw   = (const float*)d_in[6];
    const float* convw = (const float*)d_in[7];
    const float* convb = (const float*)d_in[8];
    const float* xpw   = (const float*)d_in[9];
    const float* dtw   = (const float*)d_in[10];
    const float* dtb   = (const float*)d_in[11];
    const float* alog  = (const float*)d_in[12];
    const float* dp    = (const float*)d_in[13];
    const float* outw  = (const float*)d_in[14];
    const float* n2g   = (const float*)d_in[15];
    const float* n2b   = (const float*)d_in[16];
    const float* gatew = (const float*)d_in[17];
    const float* fc1w  = (const float*)d_in[18];
    const float* fc1b  = (const float*)d_in[19];
    const float* pew   = (const float*)d_in[20];
    const float* peb   = (const float*)d_in[21];
    const float* fc2w  = (const float*)d_in[22];
    const float* fc2b  = (const float*)d_in[23];
    float* out = (float*)d_out;

    char* ws = (char*)d_ws;
    size_t off = 0;
    auto alloc = [&](size_t bytes) { size_t o = off; off = (off + bytes + 255) & ~(size_t)255; return o; };
    size_t oXN16   = alloc((size_t)M1 * CCH * 2);
    size_t oINW16  = alloc((size_t)768 * 128 * 2);
    size_t oXPW16  = alloc((size_t)NXP * DI * 2);
    size_t oDTW16  = alloc((size_t)DI * RRANK * 2);
    size_t oOUTW16 = alloc((size_t)CCH * DI * 2);
    size_t oFC1W16 = alloc((size_t)FCH * CCH * 2);
    size_t oFC2W16 = alloc((size_t)CCH * FCH * 2);
    size_t oHG     = alloc((size_t)M1 * 768 * 4);
    size_t oU      = alloc((size_t)M1 * DI * 4);
    size_t oU16    = alloc((size_t)M1 * DI * 2);
    size_t oDTBC   = alloc((size_t)M1 * NXP * 4);
    size_t oDTIN16 = alloc((size_t)M1 * RRANK * 2);
    size_t oDT     = alloc((size_t)M1 * DI * 4);
    size_t oAP     = alloc((size_t)BB * DI * NCHUNK * NST * 4);
    size_t oSC     = alloc((size_t)BB * DI * NCHUNK * NST * 4);
    size_t oINIT   = alloc((size_t)BB * DI * NCHUNK * NST * 4);
    size_t oY16    = alloc((size_t)M1 * DI * 2);
    size_t oYOUT   = alloc((size_t)M1 * CCH * 4);
    size_t oINV    = alloc((size_t)LSEQ * 4);
    size_t oGMEAN  = alloc((size_t)NBATCH * 3 * CCH * 4);
    size_t oGCOEF  = alloc((size_t)NBATCH * 4 * 4);
    size_t oX1     = alloc((size_t)M2 * CCH * 4);
    size_t oXN2    = alloc((size_t)M2 * CCH * 2);
    size_t oT1     = alloc((size_t)M2 * FCH * 4);
    size_t oT2     = alloc((size_t)M2 * FCH * 2);

    _Float16* xn16  = (_Float16*)(ws + oXN16);
    _Float16* inw16 = (_Float16*)(ws + oINW16);
    _Float16* xpw16 = (_Float16*)(ws + oXPW16);
    _Float16* dtw16 = (_Float16*)(ws + oDTW16);
    _Float16* outw16= (_Float16*)(ws + oOUTW16);
    _Float16* fc1w16= (_Float16*)(ws + oFC1W16);
    _Float16* fc2w16= (_Float16*)(ws + oFC2W16);
    float*    hg    = (float*)(ws + oHG);
    float*    u     = (float*)(ws + oU);
    _Float16* u16   = (_Float16*)(ws + oU16);
    float*    dtbc  = (float*)(ws + oDTBC);
    _Float16* dtin16= (_Float16*)(ws + oDTIN16);
    float*    dtbuf = (float*)(ws + oDT);
    float*    Ap    = (float*)(ws + oAP);
    float*    Sc    = (float*)(ws + oSC);
    float*    initS = (float*)(ws + oINIT);
    _Float16* y16   = (_Float16*)(ws + oY16);
    float*    yout  = (float*)(ws + oYOUT);
    int*      inv   = (int*)(ws + oINV);
    float*    gmean = (float*)(ws + oGMEAN);
    float*    gcoef = (float*)(ws + oGCOEF);
    float*    x1    = (float*)(ws + oX1);
    _Float16* xn2   = (_Float16*)(ws + oXN2);
    float*    t1    = (float*)(ws + oT1);
    _Float16* t2    = (_Float16*)(ws + oT2);

    // inverse permutation + weight conversions
    hipLaunchKernelGGL(k_invperm, dim3(16), dim3(256), 0, stream, ridx, inv);
    hipLaunchKernelGGL(k_f32_to_f16, dim3((768*128+255)/256),  dim3(256), 0, stream, inw,  inw16,  768*128);
    hipLaunchKernelGGL(k_f32_to_f16, dim3((NXP*DI+255)/256),   dim3(256), 0, stream, xpw,  xpw16,  NXP*DI);
    hipLaunchKernelGGL(k_f32_to_f16, dim3((DI*RRANK+255)/256), dim3(256), 0, stream, dtw,  dtw16,  DI*RRANK);
    hipLaunchKernelGGL(k_f32_to_f16, dim3((CCH*DI+255)/256),   dim3(256), 0, stream, outw, outw16, CCH*DI);
    hipLaunchKernelGGL(k_f32_to_f16, dim3((FCH*CCH+255)/256),  dim3(256), 0, stream, fc1w, fc1w16, FCH*CCH);
    hipLaunchKernelGGL(k_f32_to_f16, dim3((CCH*FCH+255)/256),  dim3(256), 0, stream, fc2w, fc2w16, CCH*FCH);

    // streams + LN1
    hipLaunchKernelGGL(k_prep_ln, dim3(M1/4), dim3(128), 0, stream, x, ridx, n1g, n1b, xn16);

    // in_proj: (24576x128) x (768x128)^T
    { int ngr = 768/64, mgr = M1/128; hipLaunchKernelGGL((k_gemm_wmma<128, EPI_NONE>), dim3(mgr*ngr), dim3(256), 0, stream,
        xn16, inw16, hg, M1, 768, 768, (const float*)nullptr, (const float*)nullptr, ngr); }

    // causal depthwise conv + silu -> u (f32 + f16)
    hipLaunchKernelGGL(k_conv_silu, dim3((M1*DI+255)/256), dim3(256), 0, stream, hg, convw, convb, u, u16);

    // x_proj: (24576x384) x (136x384)^T  (N padded to 192, guarded at store)
    { int ngr = (NXP+63)/64, mgr = M1/128; hipLaunchKernelGGL((k_gemm_wmma<384, EPI_NONE>), dim3(mgr*ngr), dim3(256), 0, stream,
        u16, xpw16, dtbc, M1, NXP, NXP, (const float*)nullptr, (const float*)nullptr, ngr); }

    // dt_proj + softplus: (24576x128) x (384x128)^T
    hipLaunchKernelGGL(k_dtcvt, dim3((M1*RRANK+255)/256), dim3(256), 0, stream, dtbc, dtin16);
    { int ngr = DI/64, mgr = M1/128; hipLaunchKernelGGL((k_gemm_wmma<128, EPI_BIAS_SOFTPLUS>), dim3(mgr*ngr), dim3(256), 0, stream,
        dtin16, dtw16, dtbuf, M1, DI, DI, dtb, (const float*)nullptr, ngr); }

    // chunked selective scan (3 passes), fused gating epilogue -> y16
    hipLaunchKernelGGL(k_scan_pass1, dim3((BB*NCHUNK*DI+255)/256), dim3(256), 0, stream, dtbuf, dtbc, u, alog, Ap, Sc);
    hipLaunchKernelGGL(k_scan_pass2, dim3((BB*DI*NST+255)/256),    dim3(256), 0, stream, Ap, Sc, initS);
    hipLaunchKernelGGL(k_scan_pass3, dim3((BB*NCHUNK*DI+255)/256), dim3(256), 0, stream, dtbuf, dtbc, u, alog, initS, hg, dp, y16);

    // out_proj: (24576x384) x (128x384)^T
    { int ngr = CCH/64, mgr = M1/128; hipLaunchKernelGGL((k_gemm_wmma<384, EPI_NONE>), dim3(mgr*ngr), dim3(256), 0, stream,
        y16, outw16, yout, M1, CCH, CCH, (const float*)nullptr, (const float*)nullptr, ngr); }

    // stream gating
    hipLaunchKernelGGL(k_gate_mean,    dim3((BB*CCH)/4), dim3(128), 0, stream, yout, gmean);
    hipLaunchKernelGGL(k_gate_softmax, dim3(NBATCH), dim3(96), 0, stream, gmean, gatew, gcoef);
    hipLaunchKernelGGL(k_combine_ln2,  dim3(M2/4), dim3(128), 0, stream, x, yout, inv, gcoef, n2g, n2b, x1, xn2);

    // MixFFN: fc1 (+bias), depthwise 3x3 + gelu, fc2 (+bias + residual) -> out
    { int ngr = FCH/64, mgr = M2/128; hipLaunchKernelGGL((k_gemm_wmma<128, EPI_BIAS>), dim3(mgr*ngr), dim3(256), 0, stream,
        xn2, fc1w16, t1, M2, FCH, FCH, fc1b, (const float*)nullptr, ngr); }
    hipLaunchKernelGGL(k_dwconv_gelu, dim3((M2*FCH+255)/256), dim3(256), 0, stream, t1, pew, peb, t2);
    { int ngr = CCH/64, mgr = M2/128; hipLaunchKernelGGL((k_gemm_wmma<512, EPI_BIAS_RESID>), dim3(mgr*ngr), dim3(256), 0, stream,
        t2, fc2w16, out, M2, CCH, CCH, fc2b, x1, ngr); }
}
